// PatchTokenMixer_70128226009806
// MI455X (gfx1250) — compile-verified, matrix operand order checked by reading, not run on hardware
//
#include <hip/hip_runtime.h>

typedef __attribute__((ext_vector_type(16))) _Float16 v16h;
typedef __attribute__((ext_vector_type(8)))  float    v8f;

#define WMMA_F16(a, b, c) \
  __builtin_amdgcn_wmma_f32_16x16x32_f16(false, (a), false, (b), (short)0, (c), false, false)

// Load one 16x32 f16 operand fragment from a row-major tile (works for LDS or
// global pointers). Layout per CDNA5 ISA 7.12.2 (16-bit A-matrix 16x32):
//   lane 0-15 : row = lane,   K = {0..7, 16..23}
//   lane 16-31: row = lane-16, K = {8..15, 24..31}
// B operand uses the symmetric layout with "row" = output column (i.e. a row
// of the row-major weight matrix W, since out = act @ W^T).
static __device__ inline v16h load_frag(const _Float16* base, int stride) {
  const int lane = threadIdx.x & 31;
  const int r    = lane & 15;
  const int kg   = lane >> 4;
  const _Float16* p = base + r * stride + kg * 8;
  v16h f;
#pragma unroll
  for (int i = 0; i < 8; ++i) f[i] = p[i];
#pragma unroll
  for (int i = 0; i < 8; ++i) f[8 + i] = p[16 + i];
  return f;
}

// Stage a 64x96 f16 tile (row stride 96 in global) into LDS with row stride
// 104 halves, using gfx1250 async global->LDS copies (ASYNCcnt domain, no
// VGPR staging). Rows are 192B = 12 x 16B chunks; both global and LDS chunk
// addresses are 16B aligned. The low 32 bits of a flat pointer to LDS are the
// wave-relative LDS byte address (ISA aperture rule), which is what the
// instruction's VDST operand wants.
static __device__ inline void stage_tile_async(_Float16* lds_base,
                                               const _Float16* gsrc, int tid) {
  for (int idx = tid; idx < 64 * 12; idx += 256) {
    int m = idx / 12, c = idx % 12;
    unsigned int lds = (unsigned int)(size_t)(lds_base + m * 104) + c * 16;
    unsigned long long g = (unsigned long long)(size_t)(gsrc + m * 96 + c * 8);
    asm volatile("global_load_async_to_lds_b128 %0, %1, off"
                 :: "v"(lds), "v"(g) : "memory");
  }
  asm volatile("s_wait_asynccnt 0x0" ::: "memory");
}

// ---------------------------------------------------------------- utilities
__global__ void f32_to_f16_kernel(const float* __restrict__ src,
                                  _Float16* __restrict__ dst, int n) {
  int i = blockIdx.x * blockDim.x + threadIdx.x;
  int stride = gridDim.x * blockDim.x;
  for (; i < n; i += stride) dst[i] = (_Float16)src[i];
}

// ------------------------------------------------------------- patch embed
// tok[b, l, e] = sum_k x_patch[l, k] * pe_w[e, k] + pe_b[e]
// M=131072 (64 tokens / block), K=128, N=96.
__global__ __launch_bounds__(256) void patch_embed_kernel(
    const float* __restrict__ x, const _Float16* __restrict__ wh,
    const float* __restrict__ peb, float* __restrict__ tok) {
  __shared__ __align__(16) _Float16 As[64 * 136];
  const int tid = threadIdx.x;
  const int t0  = blockIdx.x * 64;
  const int b   = t0 >> 14;
  const int rr  = t0 & 16383;
  const int h   = rr >> 7;
  const int w0  = rr & 127;
  __builtin_prefetch(wh, 0, 3);
  // Gather A tile: A[m][k], k = c*4 + ph*2 + pw ; src x[b][c][2h+ph][2(w0+m)+pw]
  for (int idx = tid; idx < 64 * 128; idx += 256) {
    int k = idx >> 6;
    int m = idx & 63;
    int c = k >> 2, ph = (k >> 1) & 1, pw = k & 1;
    size_t src = (((size_t)b * 32 + c) * 256 + (2 * h + ph)) * 256 + 2 * (w0 + m) + pw;
    As[m * 136 + k] = (_Float16)x[src];
  }
  __syncthreads();
  const int wid  = tid >> 5;
  const int lane = tid & 31;
  const int nloc = lane & 15;
  const int mb   = (lane >> 4) * 8;
#pragma unroll
  for (int j = 0; j < 3; ++j) {           // 4x6 = 24 tiles over 8 waves
    int t = wid * 3 + j;
    int mt = t / 6, nt = t % 6;
    v8f acc = {};
#pragma unroll
    for (int kk = 0; kk < 4; ++kk) {
      v16h a  = load_frag(As + mt * 16 * 136 + kk * 32, 136);
      v16h bb = load_frag(wh + (nt * 16) * 128 + kk * 32, 128);
      acc = WMMA_F16(a, bb, acc);
    }
    int n = nt * 16 + nloc;
    float bias = peb[n];
    float* dst = tok + (size_t)(t0 + mt * 16 + mb) * 96 + n;
#pragma unroll
    for (int r = 0; r < 8; ++r) dst[(size_t)r * 96] = acc[r] + bias;
  }
}

// -------------------------------------------------------------- layer norm
// One wave per token (96 channels = 3/lane). Optionally writes output in
// shifted+window-partitioned order (for attention) as f16.
__global__ __launch_bounds__(256) void ln_kernel(
    const float* __restrict__ tok, const float* __restrict__ g,
    const float* __restrict__ bta, _Float16* __restrict__ dst,
    int shift, int windowed) {
  const int wid = threadIdx.x >> 5, lane = threadIdx.x & 31;
  const int o  = blockIdx.x * 8 + wid;
  const int bi = o >> 14, q = o & 16383;
  int lsrc;
  if (windowed) {
    int win = q >> 6, n = q & 63;
    int wy = win >> 4, wx = win & 15, iy = n >> 3, ix = n & 7;
    int h = (wy * 8 + iy + shift) & 127;
    int w = (wx * 8 + ix + shift) & 127;
    lsrc = h * 128 + w;
  } else {
    lsrc = q;
  }
  const float* src = tok + ((size_t)bi * 16384 + lsrc) * 96;
  float v0 = src[lane], v1 = src[lane + 32], v2 = src[lane + 64];
  float s = v0 + v1 + v2, s2 = v0 * v0 + v1 * v1 + v2 * v2;
#pragma unroll
  for (int m = 16; m > 0; m >>= 1) {
    s  += __shfl_xor(s, m, 32);
    s2 += __shfl_xor(s2, m, 32);
  }
  float mu  = s * (1.0f / 96.0f);
  float var = s2 * (1.0f / 96.0f) - mu * mu;
  float inv = rsqrtf(var + 1e-5f);
  _Float16* d = dst + (size_t)o * 96;
  d[lane]      = (_Float16)((v0 - mu) * inv * g[lane]      + bta[lane]);
  d[lane + 32] = (_Float16)((v1 - mu) * inv * g[lane + 32] + bta[lane + 32]);
  d[lane + 64] = (_Float16)((v2 - mu) * inv * g[lane + 64] + bta[lane + 64]);
}

// ------------------------------------------------------------------- qkv
// Per window: (64x96) @ (288x96)^T + bias -> split into q(,*scale)/k/v with
// layout (win, head, n=64, hd=24) f16.
__global__ __launch_bounds__(256) void qkv_kernel(
    const _Float16* __restrict__ xw, const _Float16* __restrict__ wh,
    const float* __restrict__ qkvb, _Float16* __restrict__ qd,
    _Float16* __restrict__ kd, _Float16* __restrict__ vd) {
  __shared__ __align__(16) _Float16 As[64 * 104];
  const int tid = threadIdx.x;
  const int win = blockIdx.x;
  __builtin_prefetch(wh, 0, 3);
  stage_tile_async(As, xw + (size_t)win * 64 * 96, tid);
  __syncthreads();
  const int wid = tid >> 5, lane = tid & 31;
  const int nloc = lane & 15, mb = (lane >> 4) * 8;
#pragma unroll
  for (int j = 0; j < 9; ++j) {           // 4x18 = 72 tiles over 8 waves
    int t = wid * 9 + j;
    int mt = t / 18, nt = t % 18;
    v8f acc = {};
#pragma unroll
    for (int kk = 0; kk < 3; ++kk) {
      v16h a  = load_frag(As + mt * 16 * 104 + kk * 32, 104);
      v16h bb = load_frag(wh + (nt * 16) * 96 + kk * 32, 96);
      acc = WMMA_F16(a, bb, acc);
    }
    int n = nt * 16 + nloc;
    int which = n / 96;
    int head  = (n % 96) / 24;
    int dch   = n % 24;
    float bias  = qkvb[n];
    float scale = (which == 0) ? 0.2041241452319315f : 1.0f;  // hd^-0.5
    _Float16* base = (which == 0) ? qd : (which == 1) ? kd : vd;
    size_t o0 = ((size_t)(win * 4 + head) * 64 + mt * 16 + mb) * 24 + dch;
#pragma unroll
    for (int r = 0; r < 8; ++r)
      base[o0 + (size_t)r * 24] = (_Float16)((acc[r] + bias) * scale);
  }
}

// ------------------------------------------------------------- attention
static __device__ inline float attn_bias(const float* __restrict__ rpb, int head,
                                         int i, int j, int wy, int wx, int shift) {
  int yi = i >> 3, xi = i & 7, yj = j >> 3, xj = j & 7;
  float b = rpb[((yi - yj + 7) * 15 + (xi - xj + 7)) * 4 + head];
  if (shift) {
    int hi = wy * 8 + yi, wi = wx * 8 + xi;
    int hj = wy * 8 + yj, wj = wx * 8 + xj;
    int ri = ((hi < 120) ? 0 : (hi < 124) ? 1 : 2) * 3 +
             ((wi < 120) ? 0 : (wi < 124) ? 1 : 2);
    int rj = ((hj < 120) ? 0 : (hj < 124) ? 1 : 2) * 3 +
             ((wj < 120) ? 0 : (wj < 124) ? 1 : 2);
    if (ri != rj) b -= 100.0f;
  }
  return b;
}

// One block per (window, head). S = qk^T (+bias/mask), softmax, O = P v.
__global__ __launch_bounds__(256) void attn_kernel(
    const _Float16* __restrict__ qd, const _Float16* __restrict__ kd,
    const _Float16* __restrict__ vd, const float* __restrict__ rpb,
    _Float16* __restrict__ obuf, int shift) {
  __shared__ __align__(16) _Float16 Qs[64 * 32];  // K padded 24 -> 32 (zero fill)
  __shared__ __align__(16) _Float16 Ks[64 * 32];
  __shared__ __align__(16) _Float16 VTs[32 * 72]; // v^T: [d(pad32)][j=64], stride 72
  __shared__ float    Ss[64 * 68];
  __shared__ __align__(16) _Float16 Ps[64 * 72];
  const int tid = threadIdx.x;
  const int wh  = blockIdx.x;
  const int win = wh >> 2, head = wh & 3;
  const int winb = win & 255, wy = winb >> 4, wx = winb & 15;
  const _Float16* qsrc = qd + (size_t)wh * 64 * 24;
  const _Float16* ksrc = kd + (size_t)wh * 64 * 24;
  const _Float16* vsrc = vd + (size_t)wh * 64 * 24;
  for (int idx = tid; idx < 64 * 32; idx += 256) {
    int m = idx >> 5, c = idx & 31;
    _Float16 z = (_Float16)0.0f;
    Qs[idx] = (c < 24) ? qsrc[m * 24 + c] : z;
    Ks[idx] = (c < 24) ? ksrc[m * 24 + c] : z;
  }
  for (int idx = tid; idx < 32 * 64; idx += 256) {
    int d = idx >> 6, jj = idx & 63;
    VTs[d * 72 + jj] = (d < 24) ? vsrc[jj * 24 + d] : (_Float16)0.0f;
  }
  __syncthreads();
  const int wid = tid >> 5, lane = tid & 31;
  const int nloc = lane & 15, mb = (lane >> 4) * 8;
  // ---- S = Q @ K^T (4x4 tiles over 8 waves, single K-step of 32)
#pragma unroll
  for (int j = 0; j < 2; ++j) {
    int t = wid * 2 + j;
    int mt = t >> 2, nt = t & 3;
    v8f acc = {};
    v16h a  = load_frag(Qs + mt * 16 * 32, 32);
    v16h bb = load_frag(Ks + nt * 16 * 32, 32);
    acc = WMMA_F16(a, bb, acc);
#pragma unroll
    for (int r = 0; r < 8; ++r)
      Ss[(mt * 16 + mb + r) * 68 + nt * 16 + nloc] = acc[r];
  }
  __syncthreads();
  // ---- softmax rows (wave per 8 rows, lane handles cols {lane, lane+32})
#pragma unroll
  for (int rr = 0; rr < 8; ++rr) {
    int i = wid * 8 + rr;
    float x0 = Ss[i * 68 + lane]      + attn_bias(rpb, head, i, lane,      wy, wx, shift);
    float x1 = Ss[i * 68 + lane + 32] + attn_bias(rpb, head, i, lane + 32, wy, wx, shift);
    float mx = fmaxf(x0, x1);
#pragma unroll
    for (int m = 16; m > 0; m >>= 1) mx = fmaxf(mx, __shfl_xor(mx, m, 32));
    float e0 = expf(x0 - mx), e1 = expf(x1 - mx);
    float s = e0 + e1;
#pragma unroll
    for (int m = 16; m > 0; m >>= 1) s += __shfl_xor(s, m, 32);
    float inv = 1.0f / s;
    Ps[i * 72 + lane]      = (_Float16)(e0 * inv);
    Ps[i * 72 + lane + 32] = (_Float16)(e1 * inv);
  }
  __syncthreads();
  // ---- O = P @ V (4x2 tiles over 8 waves, K=64 -> 2 steps)
  {
    int mt = wid >> 1, nt = wid & 1;
    v8f acc = {};
#pragma unroll
    for (int kk = 0; kk < 2; ++kk) {
      v16h a  = load_frag(Ps + mt * 16 * 72 + kk * 32, 72);
      v16h bb = load_frag(VTs + (nt * 16) * 72 + kk * 32, 72);
      acc = WMMA_F16(a, bb, acc);
    }
    int dloc = nt * 16 + nloc;
    if (dloc < 24) {
      size_t o0 = (size_t)win * 64 * 96 + (size_t)(mt * 16 + mb) * 96 + head * 24 + dloc;
#pragma unroll
      for (int r = 0; r < 8; ++r) obuf[o0 + (size_t)r * 96] = (_Float16)acc[r];
    }
  }
}

// --------------------------------------------------- proj + window-reverse
__global__ __launch_bounds__(256) void proj_kernel(
    const _Float16* __restrict__ obuf, const _Float16* __restrict__ wh,
    const float* __restrict__ pb, float* __restrict__ tok, int shift) {
  __shared__ __align__(16) _Float16 As[64 * 104];
  const int tid = threadIdx.x;
  const int win = blockIdx.x;
  const int b = win >> 8, winb = win & 255, wy = winb >> 4, wx = winb & 15;
  __builtin_prefetch(wh, 0, 3);
  stage_tile_async(As, obuf + (size_t)win * 64 * 96, tid);
  __syncthreads();
  const int wid = tid >> 5, lane = tid & 31;
  const int nloc = lane & 15, mb = (lane >> 4) * 8;
#pragma unroll
  for (int j = 0; j < 3; ++j) {           // 4x6 = 24 tiles over 8 waves
    int t = wid * 3 + j;
    int mt = t / 6, nt = t % 6;
    v8f acc = {};
#pragma unroll
    for (int kk = 0; kk < 3; ++kk) {
      v16h a  = load_frag(As + mt * 16 * 104 + kk * 32, 104);
      v16h bb = load_frag(wh + (nt * 16) * 96 + kk * 32, 96);
      acc = WMMA_F16(a, bb, acc);
    }
    int n = nt * 16 + nloc;
    float bias = pb[n];
#pragma unroll
    for (int r = 0; r < 8; ++r) {
      int mloc = mt * 16 + mb + r;
      int h = (wy * 8 + (mloc >> 3) + shift) & 127;   // un-roll (+shift)
      int w = (wx * 8 + (mloc & 7) + shift) & 127;
      size_t ti = ((size_t)b * 16384 + h * 128 + w) * 96 + n;
      tok[ti] += acc[r] + bias;                       // residual
    }
  }
}

// ------------------------------------------------- fused FFN (GEMM-gelu-GEMM)
__global__ __launch_bounds__(256) void ffn_kernel(
    const _Float16* __restrict__ hln, const _Float16* __restrict__ w1,
    const float* __restrict__ b1, const _Float16* __restrict__ w2,
    const float* __restrict__ b2, float* __restrict__ tok) {
  __shared__ __align__(16) _Float16 As[64 * 104];
  __shared__ __align__(16) _Float16 Ys[64 * 392];
  const int tid = threadIdx.x;
  const size_t t0 = (size_t)blockIdx.x * 64;
  __builtin_prefetch(w1, 0, 3);
  __builtin_prefetch(w2, 0, 3);
  stage_tile_async(As, hln + t0 * 96, tid);
  __syncthreads();
  const int wid = tid >> 5, lane = tid & 31;
  const int nloc = lane & 15, mb = (lane >> 4) * 8;
  // ---- phase 1: Y = gelu(hln @ w1^T + b1), 4x24 tiles over 8 waves
#pragma unroll
  for (int j = 0; j < 12; ++j) {
    int t = wid * 12 + j;
    int mt = t / 24, nt = t % 24;
    v8f acc = {};
#pragma unroll
    for (int kk = 0; kk < 3; ++kk) {
      v16h a  = load_frag(As + mt * 16 * 104 + kk * 32, 104);
      v16h bb = load_frag(w1 + (nt * 16) * 96 + kk * 32, 96);
      acc = WMMA_F16(a, bb, acc);
    }
    int n = nt * 16 + nloc;
    float bias = b1[n];
#pragma unroll
    for (int r = 0; r < 8; ++r) {
      float v = acc[r] + bias;
      float gl = 0.5f * v * (1.0f + erff(v * 0.7071067811865475f));
      Ys[(mt * 16 + mb + r) * 392 + n] = (_Float16)gl;
    }
  }
  __syncthreads();
  // ---- phase 2: tok += Y @ w2^T + b2, 4x6 tiles over 8 waves, K=384
#pragma unroll
  for (int j = 0; j < 3; ++j) {
    int t = wid * 3 + j;
    int mt = t / 6, nt = t % 6;
    v8f acc = {};
#pragma unroll
    for (int kk = 0; kk < 12; ++kk) {
      v16h a  = load_frag(Ys + mt * 16 * 392 + kk * 32, 392);
      v16h bb = load_frag(w2 + (nt * 16) * 384 + kk * 32, 384);
      acc = WMMA_F16(a, bb, acc);
    }
    int n = nt * 16 + nloc;
    float bias = b2[n];
    float* dst = tok + (t0 + mt * 16 + mb) * 96 + n;
#pragma unroll
    for (int r = 0; r < 8; ++r) dst[(size_t)r * 96] += acc[r] + bias;
  }
}

// -------------------------------------------------- final back projection
__global__ __launch_bounds__(256) void back_kernel(
    const _Float16* __restrict__ hln, const _Float16* __restrict__ wh,
    const float* __restrict__ bb_, float* __restrict__ out) {
  __shared__ __align__(16) _Float16 As[64 * 104];
  const int tid = threadIdx.x;
  const int t0 = blockIdx.x * 64;
  const int b = t0 >> 14, l0 = t0 & 16383;
  __builtin_prefetch(wh, 0, 3);
  stage_tile_async(As, hln + (size_t)t0 * 96, tid);
  __syncthreads();
  const int wid = tid >> 5, lane = tid & 31;
  const int nloc = lane & 15, mb = (lane >> 4) * 8;
  // 4x2 = 8 tiles, one per wave
  int mt = wid >> 1, nt = wid & 1;
  v8f acc = {};
#pragma unroll
  for (int kk = 0; kk < 3; ++kk) {
    v16h a  = load_frag(As + mt * 16 * 104 + kk * 32, 104);
    v16h bb = load_frag(wh + (nt * 16) * 96 + kk * 32, 96);
    acc = WMMA_F16(a, bb, acc);
  }
  int n = nt * 16 + nloc;
  float bias = bb_[n];
  float* dst = out + ((size_t)b * 32 + n) * 16384 + l0 + mt * 16 + mb;
#pragma unroll
  for (int r = 0; r < 8; ++r) dst[r] = acc[r] + bias;
}

// ------------------------------------------------------------------ driver
extern "C" void kernel_launch(void* const* d_in, const int* in_sizes, int n_in,
                              void* d_out, int out_size, void* d_ws, size_t ws_size,
                              hipStream_t stream) {
  (void)in_sizes; (void)n_in; (void)out_size; (void)ws_size;
  // Workspace layout (bytes):
  char* ws = (char*)d_ws;
  float*    tok = (float*)ws;                          // 131072*96*4 = 50331648
  _Float16* xw  = (_Float16*)(ws + 50331648ull);       // 131072*96*2 = 25165824 (also hln)
  _Float16* qb  = (_Float16*)(ws + 75497472ull);       // 2048*4*64*24*2
  _Float16* kb  = (_Float16*)(ws + 100663296ull);
  _Float16* vb  = (_Float16*)(ws + 125829120ull);
  _Float16* ob  = (_Float16*)(ws + 150994944ull);
  char* wp = ws + 176160768ull;                        // f16 weight copies
  _Float16* pe_w_h = (_Float16*)wp; wp += 96 * 128 * 2;
  _Float16 *qkvw_h[2], *pw_h[2], *f1w_h[2], *f2w_h[2];
  for (int blk = 0; blk < 2; ++blk) {
    qkvw_h[blk] = (_Float16*)wp; wp += 288 * 96 * 2;
    pw_h[blk]   = (_Float16*)wp; wp += 96 * 96 * 2;
    f1w_h[blk]  = (_Float16*)wp; wp += 384 * 96 * 2;
    f2w_h[blk]  = (_Float16*)wp; wp += 96 * 384 * 2;
  }
  _Float16* back_w_h = (_Float16*)wp;

  auto cvt = [&](const void* s, int n, _Float16* d) {
    f32_to_f16_kernel<<<(n + 255) / 256, 256, 0, stream>>>((const float*)s, d, n);
  };
  cvt(d_in[1], 96 * 128, pe_w_h);
  for (int blk = 0; blk < 2; ++blk) {
    int base = 3 + blk * 13;
    cvt(d_in[base + 2],  288 * 96, qkvw_h[blk]);
    cvt(d_in[base + 4],  96 * 96,  pw_h[blk]);
    cvt(d_in[base + 9],  384 * 96, f1w_h[blk]);
    cvt(d_in[base + 11], 96 * 384, f2w_h[blk]);
  }
  cvt(d_in[31], 32 * 96, back_w_h);

  patch_embed_kernel<<<2048, 256, 0, stream>>>(
      (const float*)d_in[0], pe_w_h, (const float*)d_in[2], tok);

  for (int blk = 0; blk < 2; ++blk) {
    int base  = 3 + blk * 13;
    int shift = blk ? 4 : 0;
    ln_kernel<<<16384, 256, 0, stream>>>(
        tok, (const float*)d_in[base + 0], (const float*)d_in[base + 1], xw, shift, 1);
    qkv_kernel<<<2048, 256, 0, stream>>>(
        xw, qkvw_h[blk], (const float*)d_in[base + 3], qb, kb, vb);
    attn_kernel<<<8192, 256, 0, stream>>>(
        qb, kb, vb, (const float*)d_in[base + 6], ob, shift);
    proj_kernel<<<2048, 256, 0, stream>>>(
        ob, pw_h[blk], (const float*)d_in[base + 5], tok, shift);
    ln_kernel<<<16384, 256, 0, stream>>>(
        tok, (const float*)d_in[base + 7], (const float*)d_in[base + 8], xw, 0, 0);
    ffn_kernel<<<2048, 256, 0, stream>>>(
        xw, f1w_h[blk], (const float*)d_in[base + 10],
        f2w_h[blk], (const float*)d_in[base + 12], tok);
  }
  ln_kernel<<<16384, 256, 0, stream>>>(
      tok, (const float*)d_in[29], (const float*)d_in[30], xw, 0, 0);
  back_kernel<<<2048, 256, 0, stream>>>(
      xw, back_w_h, (const float*)d_in[32], (float*)d_out);
}
